// SelfSimilarity_72627896975436
// MI455X (gfx1250) — matrix operand order repaired and checked by reading, not compiled, not run on hardware
//
#include <hip/hip_runtime.h>

// ---------------------------------------------------------------------------
// Self-similarity softmax for MI455X (gfx1250), wave32 + WMMA bf16 split path.
//   x: [8, 2048, 128] fp32  ->  out: [8, 2048, 2048, 1] fp32
//   dist[b,i,j] = max(||xi||^2 - 2 xi.xj + ||xj||^2, 0)
//   out = softmax_j(-T * dist)    (row max logit is exactly 0 -> no online max)
//
// Kernel 1 (prep): x -> bf16 hi/lo split (xhi, xlo) + fp32 row norms, in d_ws.
//   dot ~= hi*hi + hi*lo + lo*hi  (~fp32 accuracy, 3 WMMAs per K=32 chunk)
// Kernel 2 (main): per block: 64 i-rows x full j sweep, two passes (row sums,
//   then normalize + NT store). B tiles stream into double-buffered LDS via
//   CDNA5 GLOBAL_LOAD_ASYNC_TO_LDS_B128 (ASYNCcnt), overlapping the WMMAs.
//   A fragments live in registers, loaded straight from L2-resident bf16.
// Workspace: 4 MB xhi + 4 MB xlo + 64 KB sq  (8.44 MB of d_ws).
// ---------------------------------------------------------------------------

typedef __attribute__((ext_vector_type(16))) __bf16 v16bf;
typedef __attribute__((ext_vector_type(8)))  float  v8f;

#define NBATCH 8
#define NPTS   2048
#define NDIM   128
#define ROWS   64                 // i-rows per workgroup
#define JT     128                // j-columns per staged tile
#define NITER  (NPTS / JT)        // 16
#define THREADS 512               // 16 wave32
#define TEMPERATURE (1.0f / 13.544f)
// exp(-T*d) == exp2(K2F*d)
#define K2F (-1.44269504088896340736f * TEMPERATURE)

// d_ws layout (bytes)
#define XLO_OFF ((size_t)NBATCH * NPTS * NDIM * 2)   // 4 MB
#define SQ_OFF  ((size_t)NBATCH * NPTS * NDIM * 4)   // 8 MB

union FragU { uint4 q[2]; v16bf v; };
union Pack8 { __bf16 e[8]; uint4 q; };

// A operand (16x32 bf16, ISA 7.12.2): lane L holds row (L&15); K elements
// [c0, c0+8) in VGPR0-3 and [c0+16, c0+24) in VGPR4-7, c0 = (L>>4)*8.
// Works for both global and LDS pointers (generic addressing).
static __device__ __forceinline__ v16bf load_fragA(const __bf16* buf, int row,
                                                   int kbase, int c0) {
  FragU f;
  const __bf16* p = buf + (size_t)row * NDIM + kbase + c0;
  f.q[0] = *(const uint4*)(p);
  f.q[1] = *(const uint4*)(p + 16);
  return f.v;
}

// B operand (32x16 bf16): lane L holds column (L&15); K elements [c0, c0+8)
// and [c0+16, c0+24), c0 = (L>>4)*8 (caller folds c0 into kbase).
static __device__ __forceinline__ v16bf load_fragB(const __bf16* buf, int row,
                                                   int kbase) {
  FragU f;
  const __bf16* p = buf + (size_t)row * NDIM + kbase;
  f.q[0] = *(const uint4*)(p);
  f.q[1] = *(const uint4*)(p + 16);
  return f.v;
}

static __device__ __forceinline__ v8f wmma_acc(v16bf a, v16bf b, v8f c) {
  return __builtin_amdgcn_wmma_f32_16x16x32_bf16(false, a, false, b,
                                                 (short)0, c, false, false);
}

// ======================= kernel 1: bf16 split + norms ========================
__global__ __launch_bounds__(256, 1)
void prep_bf16(const float* __restrict__ x, __bf16* __restrict__ xhi,
               __bf16* __restrict__ xlo, float* __restrict__ sq) {
  const int row = blockIdx.x * 256 + threadIdx.x;     // 16384 rows total
  const float4* src = (const float4*)(x + (size_t)row * NDIM);
  __bf16* dh = xhi + (size_t)row * NDIM;
  __bf16* dl = xlo + (size_t)row * NDIM;
  float s = 0.0f;
#pragma unroll
  for (int c = 0; c < 16; ++c) {                      // 8 elements per packet
    Pack8 hp, lp;
#pragma unroll
    for (int q = 0; q < 2; ++q) {
      float4 v = src[c * 2 + q];
      float vv[4] = {v.x, v.y, v.z, v.w};
#pragma unroll
      for (int u = 0; u < 4; ++u) {
        float f  = vv[u];
        __bf16 h = (__bf16)f;
        hp.e[q * 4 + u] = h;
        lp.e[q * 4 + u] = (__bf16)(f - (float)h);
        s += f * f;
      }
    }
    *(uint4*)(dh + c * 8) = hp.q;
    *(uint4*)(dl + c * 8) = lp.q;
  }
  sq[row] = s;
}

// ======================= kernel 2: WMMA + softmax ============================
__global__ __launch_bounds__(THREADS, 1)
void selfsim_wmma(const __bf16* __restrict__ xhi, const __bf16* __restrict__ xlo,
                  const float* __restrict__ sq, float* __restrict__ out) {
  __shared__ __align__(32) __bf16 Bhi[2][JT * NDIM];  // 2 x 32 KB double buffer
  __shared__ __align__(32) __bf16 Blo[2][JT * NDIM];  // 2 x 32 KB
  __shared__ float rowsum[ROWS];

  const int t     = threadIdx.x;
  const int wave  = t >> 5;
  const int lane  = t & 31;
  const int lhalf = lane >> 4;     // 0 | 1 (half-wave)
  const int lrow  = lane & 15;
  const int c0    = lhalf * 8;

  const int mt  = wave & 3;        // 16-row tile of the 64-row block (0..3)
  const int nt0 = wave >> 2;       // first 16-col tile (0..3)
  const int nt1 = nt0 + 4;         // second 16-col tile (4..7)

  const int b   = blockIdx.y;
  const int ib0 = blockIdx.x * ROWS;
  const size_t bq = (size_t)b * NPTS;

  if (t < ROWS) rowsum[t] = 0.0f;

  // ---- A fragments + row norms straight from global (L2-resident bf16) ----
  v16bf fah[4], fal[4];
  const int arow = ib0 + mt * 16 + lrow;
  const __bf16* xhib = xhi + bq * NDIM;
  const __bf16* xlob = xlo + bq * NDIM;
#pragma unroll
  for (int kc = 0; kc < 4; ++kc) {
    fah[kc] = load_fragA(xhib, arow, kc * 32, c0);
    fal[kc] = load_fragA(xlob, arow, kc * 32, c0);
  }
  float sa[8];
#pragma unroll
  for (int e = 0; e < 8; ++e) sa[e] = sq[bq + ib0 + mt * 16 + e + 8 * lhalf];
  __syncthreads();

  // ---- async B-tile copy: 64 KB contiguous, 128 B per thread --------------
  const unsigned voff = (unsigned)t * 64u;            // 512 thr * 64 B = 32 KB
  auto issue_copy = [&](int buf, int j0) {
    unsigned long long srcH =
        (unsigned long long)(uintptr_t)(xhib + (size_t)j0 * NDIM);
    unsigned long long srcL =
        (unsigned long long)(uintptr_t)(xlob + (size_t)j0 * NDIM);
    unsigned ldsH = (unsigned)(uintptr_t)(&Bhi[buf][0]) + voff;
    unsigned ldsL = (unsigned)(uintptr_t)(&Blo[buf][0]) + voff;
    // VGLOBAL async: INST_OFFSET is added to both the LDS and global address.
    asm volatile("global_load_async_to_lds_b128 %0, %1, %2 offset:0"
                 :: "v"(ldsH), "v"(voff), "s"(srcH) : "memory");
    asm volatile("global_load_async_to_lds_b128 %0, %1, %2 offset:16"
                 :: "v"(ldsH), "v"(voff), "s"(srcH) : "memory");
    asm volatile("global_load_async_to_lds_b128 %0, %1, %2 offset:32"
                 :: "v"(ldsH), "v"(voff), "s"(srcH) : "memory");
    asm volatile("global_load_async_to_lds_b128 %0, %1, %2 offset:48"
                 :: "v"(ldsH), "v"(voff), "s"(srcH) : "memory");
    asm volatile("global_load_async_to_lds_b128 %0, %1, %2 offset:0"
                 :: "v"(ldsL), "v"(voff), "s"(srcL) : "memory");
    asm volatile("global_load_async_to_lds_b128 %0, %1, %2 offset:16"
                 :: "v"(ldsL), "v"(voff), "s"(srcL) : "memory");
    asm volatile("global_load_async_to_lds_b128 %0, %1, %2 offset:32"
                 :: "v"(ldsL), "v"(voff), "s"(srcL) : "memory");
    asm volatile("global_load_async_to_lds_b128 %0, %1, %2 offset:48"
                 :: "v"(ldsL), "v"(voff), "s"(srcL) : "memory");
  };

  // ---- two 16x16 tiles per wave, A fragments reused ------------------------
  auto gemm2 = [&](int buf, v8f& a0, v8f& a1) {
    a0 = (v8f){};
    a1 = (v8f){};
    const __bf16* bh = &Bhi[buf][0];
    const __bf16* bl = &Blo[buf][0];
#pragma unroll
    for (int kc = 0; kc < 4; ++kc) {
      const int kb = kc * 32 + lhalf * 8;
      v16bf b0h = load_fragB(bh, nt0 * 16 + lrow, kb);
      v16bf b0l = load_fragB(bl, nt0 * 16 + lrow, kb);
      v16bf b1h = load_fragB(bh, nt1 * 16 + lrow, kb);
      v16bf b1l = load_fragB(bl, nt1 * 16 + lrow, kb);
      a0 = wmma_acc(fah[kc], b0h, a0);
      a0 = wmma_acc(fal[kc], b0h, a0);
      a0 = wmma_acc(fah[kc], b0l, a0);
      a1 = wmma_acc(fah[kc], b1h, a1);
      a1 = wmma_acc(fal[kc], b1h, a1);
      a1 = wmma_acc(fah[kc], b1l, a1);
    }
  };

  // ======================= pass 1: row sums ================================
  float priv[8];
#pragma unroll
  for (int e = 0; e < 8; ++e) priv[e] = 0.0f;

  issue_copy(0, 0);
  for (int it = 0; it < NITER; ++it) {
    const int cur = it & 1;
    const int j0  = it * JT;
    asm volatile("s_wait_asynccnt 0x0" ::: "memory");   // my copies landed
    __syncthreads();                                    // everyone's landed
    if (it + 1 < NITER) issue_copy(cur ^ 1, j0 + JT);   // overlap next DMA

    v8f a0, a1;
    gemm2(cur, a0, a1);

    const int jl0 = nt0 * 16 + lrow;
    const int jl1 = nt1 * 16 + lrow;
    const float sb0 = sq[bq + j0 + jl0];
    const float sb1 = sq[bq + j0 + jl1];
#pragma unroll
    for (int e = 0; e < 8; ++e) {
      float d0v = fmaxf(fmaf(-2.0f, a0[e], sa[e] + sb0), 0.0f);
      float d1v = fmaxf(fmaf(-2.0f, a1[e], sa[e] + sb1), 0.0f);
      priv[e] += exp2f(K2F * d0v) + exp2f(K2F * d1v);
    }
    __syncthreads();                                    // tile consumed
  }
#pragma unroll
  for (int e = 0; e < 8; ++e)
    atomicAdd(&rowsum[mt * 16 + e + 8 * lhalf], priv[e]);
  __syncthreads();                                      // sums complete

  float rinv[8];
#pragma unroll
  for (int e = 0; e < 8; ++e) rinv[e] = 1.0f / rowsum[mt * 16 + e + 8 * lhalf];

  // ======================= pass 2: normalize + store =======================
  issue_copy(0, 0);
  for (int it = 0; it < NITER; ++it) {
    const int cur = it & 1;
    const int j0  = it * JT;
    asm volatile("s_wait_asynccnt 0x0" ::: "memory");
    __syncthreads();
    if (it + 1 < NITER) issue_copy(cur ^ 1, j0 + JT);

    v8f a0, a1;
    gemm2(cur, a0, a1);

    const int jl0 = nt0 * 16 + lrow;
    const int jl1 = nt1 * 16 + lrow;
    const float sb0 = sq[bq + j0 + jl0];
    const float sb1 = sq[bq + j0 + jl1];
#pragma unroll
    for (int e = 0; e < 8; ++e) {
      const int il = mt * 16 + e + 8 * lhalf;          // C: M = e + 8*(lane>=16)
      const size_t orow = ((size_t)b * NPTS + (ib0 + il)) * (size_t)NPTS;
      float d0v = fmaxf(fmaf(-2.0f, a0[e], sa[e] + sb0), 0.0f);
      float d1v = fmaxf(fmaf(-2.0f, a1[e], sa[e] + sb1), 0.0f);
      float p0 = exp2f(K2F * d0v) * rinv[e];
      float p1 = exp2f(K2F * d1v) * rinv[e];
      // write-once output: non-temporal, keep L2 for x
      __builtin_nontemporal_store(p0, &out[orow + (size_t)(j0 + jl0)]);
      __builtin_nontemporal_store(p1, &out[orow + (size_t)(j0 + jl1)]);
    }
    __syncthreads();                                    // tile consumed
  }
}

extern "C" void kernel_launch(void* const* d_in, const int* in_sizes, int n_in,
                              void* d_out, int out_size, void* d_ws, size_t ws_size,
                              hipStream_t stream) {
  (void)in_sizes; (void)n_in; (void)out_size; (void)ws_size;
  const float* x = (const float*)d_in[0];   // [8, 2048, 128] fp32
  // d_in[1] (is_batch_mode) is unused by the reference computation.
  float* out = (float*)d_out;               // [8, 2048, 2048, 1] fp32

  char* ws = (char*)d_ws;                   // needs 8.44 MB scratch
  __bf16* xhi = (__bf16*)(ws);
  __bf16* xlo = (__bf16*)(ws + XLO_OFF);
  float*  sqv = (float*)(ws + SQ_OFF);

  prep_bf16<<<dim3((NBATCH * NPTS) / 256), dim3(256), 0, stream>>>(x, xhi, xlo, sqv);
  dim3 grid(NPTS / ROWS, NBATCH);           // 32 row-blocks x 8 batches
  selfsim_wmma<<<grid, dim3(THREADS), 0, stream>>>(xhi, xlo, sqv, out);
}
